// CausalSelfAttention_35536559407322
// MI455X (gfx1250) — matrix operand order repaired and checked by reading, not compiled, not run on hardware
//
#include <hip/hip_runtime.h>

// Problem constants (B=2, T=2048, C=1024, H=16, D=64, P=2048)
#define BB 2
#define TT 2048
#define CC 1024
#define HH 16
#define DD 64
#define PP 2048
#define PT 4096      // P + T
#define NQ 3072      // 3*C
#define MROWS 4096   // B*T

typedef __attribute__((ext_vector_type(16))) __bf16 v16bf;
typedef __attribute__((ext_vector_type(8)))  float  v8f;
typedef __attribute__((ext_vector_type(4)))  unsigned int u32x4;

union Frag {
  v16bf bf;
  u32x4 u4[2];
  __bf16 e[16];
};

__device__ __forceinline__ v8f wmma_bf16(const Frag& a, const Frag& b, v8f c) {
  // D = A(16x32 bf16) * B(32x16 bf16) + C(16x16 f32)
  return __builtin_amdgcn_wmma_f32_16x16x32_bf16(false, a.bf, false, b.bf,
                                                 (short)0, c, false, false);
}

// ---------------- elementwise f32 -> bf16 ----------------
__global__ void __launch_bounds__(256)
k_cvt_bf16(const float* __restrict__ in, __bf16* __restrict__ out, int n) {
  int i = (blockIdx.x * 256 + threadIdx.x) * 4;
  if (i + 3 < n) {
    float4 v = *(const float4*)(in + i);
    out[i + 0] = (__bf16)v.x;
    out[i + 1] = (__bf16)v.y;
    out[i + 2] = (__bf16)v.z;
    out[i + 3] = (__bf16)v.w;
  }
}

// ------- pack weight (K x N, row-major f32) into WMMA B-fragment order -------
// out[((kt*(N/16)+nt)*32 + lane)*16 + e] = bf16(W[(kt*32 + (lane/16)*16 + e)*N + nt*16 + lane%16])
__global__ void __launch_bounds__(256)
k_pack_w(const float* __restrict__ W, __bf16* __restrict__ out, int K, int N) {
  int tid  = blockIdx.x * 256 + threadIdx.x;
  int lane = tid & 31;
  int frag = tid >> 5;
  int ntiles = N >> 4;
  int nt = frag % ntiles;
  int kt = frag / ntiles;
  if (kt >= (K >> 5)) return;
  int half = lane >> 4;
  int n = nt * 16 + (lane & 15);
  int kbase = kt * 32 + half * 16;
  alignas(16) __bf16 tmp[16];
  #pragma unroll
  for (int e = 0; e < 16; ++e)
    tmp[e] = (__bf16)W[(size_t)(kbase + e) * N + n];
  __bf16* dst = out + ((size_t)frag * 32 + lane) * 16;
  *(u32x4*)(dst)     = *(const u32x4*)(tmp);
  *(u32x4*)(dst + 8) = *(const u32x4*)(tmp + 8);
}

// ------- convert caches: cache_k -> Kall[:, :, 0:P, :], cache_v -> Vt[:, :, :, 0:P] -------
__global__ void __launch_bounds__(256)
k_cvt_cache(const float* __restrict__ ck, const float* __restrict__ cv,
            __bf16* __restrict__ kall, __bf16* __restrict__ vt) {
  size_t i = (size_t)blockIdx.x * 256 + threadIdx.x;  // [0, B*H*P*D)
  int d  = (int)(i & 63);
  int p  = (int)((i >> 6) & 2047);
  int bh = (int)(i >> 17);
  kall[((size_t)bh * PT + p) * DD + d]  = (__bf16)ck[i];
  vt[((size_t)bh * DD + d) * PT + p]    = (__bf16)cv[i];
}

// ---- double-buffered GEMM mainloop helpers (compile-time buffer indices) ----
#define GEMM_LOAD_STEP(kt_, buf_, BNT_) do {                                   \
    a[buf_][0].u4[0] = *(const u32x4*)(arow0 + (kt_) * 32 + half * 16);        \
    a[buf_][0].u4[1] = *(const u32x4*)(arow0 + (kt_) * 32 + half * 16 + 8);    \
    a[buf_][1].u4[0] = *(const u32x4*)(arow1 + (kt_) * 32 + half * 16);        \
    a[buf_][1].u4[1] = *(const u32x4*)(arow1 + (kt_) * 32 + half * 16 + 8);    \
    const __bf16* bbase_ =                                                     \
        Bp + (((size_t)(kt_) * (BNT_) + ng * 4) * 32 + lane) * 16;             \
    _Pragma("unroll")                                                          \
    for (int nt_ = 0; nt_ < 4; ++nt_) {                                        \
      b[buf_][nt_].u4[0] = *(const u32x4*)(bbase_ + (size_t)nt_ * 512);        \
      b[buf_][nt_].u4[1] = *(const u32x4*)(bbase_ + (size_t)nt_ * 512 + 8);    \
    }                                                                          \
  } while (0)

#define GEMM_COMPUTE(buf_) do {                                                \
    _Pragma("unroll")                                                          \
    for (int nt_ = 0; nt_ < 4; ++nt_) {                                        \
      acc[0][nt_] = wmma_bf16(a[buf_][0], b[buf_][nt_], acc[0][nt_]);          \
      acc[1][nt_] = wmma_bf16(a[buf_][1], b[buf_][nt_], acc[1][nt_]);          \
    }                                                                          \
  } while (0)

// ---------------- QKV projection GEMM: (4096x1024) x (1024x3072) ----------------
// One wave computes a 32x64 tile; loads for k-step i+1 are in flight while the
// 8 WMMAs of k-step i execute (two-deep software pipeline).
__global__ void __launch_bounds__(256)
k_gemm_qkv(const __bf16* __restrict__ A, const __bf16* __restrict__ Bp,
           __bf16* __restrict__ qbuf, __bf16* __restrict__ kall,
           __bf16* __restrict__ vt) {
  int wid  = blockIdx.x * 8 + (threadIdx.x >> 5);  // 6144 waves
  int lane = threadIdx.x & 31;
  int ng = wid % 48;   // 48 groups of 64 cols
  int mt = wid / 48;   // 128 m-tiles of 32 rows
  if (mt >= 128) return;
  int half = lane >> 4, lr = lane & 15;

  const __bf16* arow0 = A + (size_t)(mt * 32 + lr) * CC;
  const __bf16* arow1 = arow0 + (size_t)16 * CC;
  v8f acc[2][4] = {};
  Frag a[2][2], b[2][4];
  GEMM_LOAD_STEP(0, 0, 192);
  for (int kt = 0; kt < 32; kt += 2) {
    GEMM_LOAD_STEP(kt + 1, 1, 192);
    GEMM_COMPUTE(0);
    if (kt + 2 < 32) GEMM_LOAD_STEP(kt + 2, 0, 192);
    GEMM_COMPUTE(1);
  }
  int part = ng >> 4;   // 0=Q 1=K 2=V
  int hh = ng & 15;     // each 64-col group == one head
  #pragma unroll
  for (int mi = 0; mi < 2; ++mi) {
    #pragma unroll
    for (int nt = 0; nt < 4; ++nt) {
      int d = nt * 16 + lr;
      #pragma unroll
      for (int r = 0; r < 8; ++r) {
        int m = r + half * 8;
        int g = mt * 32 + mi * 16 + m;
        int b_ = g >> 11, t = g & 2047;
        int bh = b_ * HH + hh;
        float val = acc[mi][nt][r];
        if (part == 0) {
          qbuf[((size_t)bh * TT + t) * DD + d] = (__bf16)(val * 0.125f);  // 1/sqrt(64)
        } else if (part == 1) {
          kall[((size_t)bh * PT + (PP + t)) * DD + d] = (__bf16)val;
        } else {
          vt[((size_t)bh * DD + d) * PT + (PP + t)] = (__bf16)val;
        }
      }
    }
  }
}

// ---------------- Flash attention step (BUF = compile-time K buffer) ----------------
template <int BUF>
__device__ __forceinline__ void attn_step(
    int j, int nkb, int qb, int w, int half, int lr,
    const __bf16* __restrict__ kb_bh, const __bf16* __restrict__ vb_bh,
    const Frag (&aq)[2], Frag (&bk)[2][2][2],
    v8f (&accO)[4], float (&mrun)[8], float (&lrun)[8],
    __bf16 (&ldsP)[8][16][32]) {
  int keybase = j * 32;

  // V fragments for this block (consumed only after softmax -> latency hidden)
  Frag bv[4];
  #pragma unroll
  for (int nt = 0; nt < 4; ++nt) {
    const __bf16* vrow = vb_bh + (size_t)(nt * 16 + lr) * PT + keybase + half * 16;
    bv[nt].u4[0] = *(const u32x4*)(vrow);
    bv[nt].u4[1] = *(const u32x4*)(vrow + 8);
  }
  __builtin_prefetch(vb_bh + (size_t)(threadIdx.x & 31) * PT + keybase + 32, 0, 1);

  // prefetch NEXT K block into the other buffer (double buffering)
  if (j + 1 < nkb) {
    int kb2 = keybase + 32;
    #pragma unroll
    for (int ntile = 0; ntile < 2; ++ntile) {
      const __bf16* krow = kb_bh + (size_t)(kb2 + ntile * 16 + lr) * DD;
      #pragma unroll
      for (int ks = 0; ks < 2; ++ks) {
        bk[1 - BUF][ntile][ks].u4[0] = *(const u32x4*)(krow + ks * 32 + half * 16);
        bk[1 - BUF][ntile][ks].u4[1] = *(const u32x4*)(krow + ks * 32 + half * 16 + 8);
      }
    }
  }

  // S = Q @ K^T from the CURRENT buffer (its loads were issued one step ago)
  v8f s[2] = {};
  #pragma unroll
  for (int ntile = 0; ntile < 2; ++ntile) {
    s[ntile] = wmma_bf16(aq[0], bk[BUF][ntile][0], s[ntile]);
    s[ntile] = wmma_bf16(aq[1], bk[BUF][ntile][1], s[ntile]);
  }

  // causal mask (only blocks overlapping the new-token region)
  if (keybase + 32 > PP) {
    #pragma unroll
    for (int ntile = 0; ntile < 2; ++ntile) {
      int ktok = keybase + ntile * 16 + lr - PP;
      #pragma unroll
      for (int r = 0; r < 8; ++r) {
        int qtok = qb * 16 + r + half * 8;
        if (ktok > qtok) s[ntile][r] = -1e30f;
      }
    }
  }

  // online softmax (row m = r + 8*half lives in one 16-lane group)
  float corr[8];
  #pragma unroll
  for (int r = 0; r < 8; ++r) {
    float tmax = fmaxf(s[0][r], s[1][r]);
    tmax = fmaxf(tmax, __shfl_xor(tmax, 1, 16));
    tmax = fmaxf(tmax, __shfl_xor(tmax, 2, 16));
    tmax = fmaxf(tmax, __shfl_xor(tmax, 4, 16));
    tmax = fmaxf(tmax, __shfl_xor(tmax, 8, 16));
    float mnew = fmaxf(mrun[r], tmax);
    corr[r] = __expf(mrun[r] - mnew);
    mrun[r] = mnew;
    float p0 = __expf(s[0][r] - mnew);
    float p1 = __expf(s[1][r] - mnew);
    s[0][r] = p0; s[1][r] = p1;
    float rs = p0 + p1;
    rs += __shfl_xor(rs, 1, 16);
    rs += __shfl_xor(rs, 2, 16);
    rs += __shfl_xor(rs, 4, 16);
    rs += __shfl_xor(rs, 8, 16);
    lrun[r] = lrun[r] * corr[r] + rs;
  }
  #pragma unroll
  for (int nt = 0; nt < 4; ++nt)
    #pragma unroll
    for (int r = 0; r < 8; ++r) accO[nt][r] *= corr[r];

  // P: C-layout -> A-layout via per-wave LDS tile (DS ops in-order per wave)
  #pragma unroll
  for (int r = 0; r < 8; ++r) {
    int m = r + half * 8;
    ldsP[w][m][lr]      = (__bf16)s[0][r];
    ldsP[w][m][16 + lr] = (__bf16)s[1][r];
  }
  Frag ap;
  ap.u4[0] = *(const u32x4*)&ldsP[w][lr][half * 16];
  ap.u4[1] = *(const u32x4*)&ldsP[w][lr][half * 16 + 8];

  // O += P @ V (V fragments already resident)
  #pragma unroll
  for (int nt = 0; nt < 4; ++nt)
    accO[nt] = wmma_bf16(ap, bv[nt], accO[nt]);
}

// ---------------- Flash attention: one wave per 16-row Q block ----------------
__global__ void __launch_bounds__(256)
k_attn(const __bf16* __restrict__ qbuf, const __bf16* __restrict__ kall,
       const __bf16* __restrict__ vt, __bf16* __restrict__ ybf) {
  __shared__ alignas(16) __bf16 ldsP[8][16][32];   // per-wave P tile
  int w    = threadIdx.x >> 5;
  int wid  = blockIdx.x * 8 + w;
  int lane = threadIdx.x & 31;
  int qb = wid & 127;     // 128 q-blocks of 16 rows
  int bh = wid >> 7;      // 0..31 (b*H + h)
  int half = lane >> 4, lr = lane & 15;

  // Q A-fragments (pre-scaled), two k-steps over D=64
  const __bf16* qrow = qbuf + ((size_t)bh * TT + qb * 16 + lr) * DD;
  Frag aq[2];
  aq[0].u4[0] = *(const u32x4*)(qrow + half * 16);
  aq[0].u4[1] = *(const u32x4*)(qrow + half * 16 + 8);
  aq[1].u4[0] = *(const u32x4*)(qrow + 32 + half * 16);
  aq[1].u4[1] = *(const u32x4*)(qrow + 40 + half * 16);

  v8f accO[4] = {};
  float mrun[8], lrun[8];
  #pragma unroll
  for (int r = 0; r < 8; ++r) { mrun[r] = -1e30f; lrun[r] = 0.f; }

  int keymax = PP + qb * 16 + 16;          // exclusive causal key limit
  int nkb = (keymax + 31) >> 5;            // 32-key blocks (>= 65)
  const __bf16* kb_bh = kall + (size_t)bh * PT * DD;
  const __bf16* vb_bh = vt + (size_t)bh * DD * PT;

  // preload K block 0 into buffer 0
  Frag bk[2][2][2];
  #pragma unroll
  for (int ntile = 0; ntile < 2; ++ntile) {
    const __bf16* krow = kb_bh + (size_t)(ntile * 16 + lr) * DD;
    #pragma unroll
    for (int ks = 0; ks < 2; ++ks) {
      bk[0][ntile][ks].u4[0] = *(const u32x4*)(krow + ks * 32 + half * 16);
      bk[0][ntile][ks].u4[1] = *(const u32x4*)(krow + ks * 32 + half * 16 + 8);
    }
  }

  int j = 0;
  for (;;) {
    attn_step<0>(j, nkb, qb, w, half, lr, kb_bh, vb_bh, aq, bk, accO, mrun, lrun, ldsP);
    if (++j >= nkb) break;
    attn_step<1>(j, nkb, qb, w, half, lr, kb_bh, vb_bh, aq, bk, accO, mrun, lrun, ldsP);
    if (++j >= nkb) break;
  }

  // normalize + store y in (B,T,C) bf16 for the out-projection
  int h = bh & 15, b_ = bh >> 4;
  #pragma unroll
  for (int r = 0; r < 8; ++r) {
    float inv = 1.f / lrun[r];
    int m = r + half * 8;
    size_t rowoff = ((size_t)(b_ * TT) + qb * 16 + m) * CC + h * DD;
    #pragma unroll
    for (int nt = 0; nt < 4; ++nt)
      ybf[rowoff + nt * 16 + lr] = (__bf16)(accO[nt][r] * inv);
  }
}

// ---------------- output projection GEMM: (4096x1024) x (1024x1024) -> f32 ----------------
__global__ void __launch_bounds__(256)
k_gemm_proj(const __bf16* __restrict__ A, const __bf16* __restrict__ Bp,
            float* __restrict__ out) {
  int wid  = blockIdx.x * 8 + (threadIdx.x >> 5);  // 2048 waves
  int lane = threadIdx.x & 31;
  int ng = wid & 15;   // 16 groups of 64 cols
  int mt = wid >> 4;   // 128 m-tiles of 32 rows
  if (mt >= 128) return;
  int half = lane >> 4, lr = lane & 15;

  const __bf16* arow0 = A + (size_t)(mt * 32 + lr) * CC;
  const __bf16* arow1 = arow0 + (size_t)16 * CC;
  v8f acc[2][4] = {};
  Frag a[2][2], b[2][4];
  GEMM_LOAD_STEP(0, 0, 64);
  for (int kt = 0; kt < 32; kt += 2) {
    GEMM_LOAD_STEP(kt + 1, 1, 64);
    GEMM_COMPUTE(0);
    if (kt + 2 < 32) GEMM_LOAD_STEP(kt + 2, 0, 64);
    GEMM_COMPUTE(1);
  }
  #pragma unroll
  for (int mi = 0; mi < 2; ++mi) {
    #pragma unroll
    for (int nt = 0; nt < 4; ++nt) {
      #pragma unroll
      for (int r = 0; r < 8; ++r) {
        int m = r + half * 8;
        out[(size_t)(mt * 32 + mi * 16 + m) * CC + ng * 64 + nt * 16 + lr] =
            acc[mi][nt][r];
      }
    }
  }
}

extern "C" void kernel_launch(void* const* d_in, const int* in_sizes, int n_in,
                              void* d_out, int out_size, void* d_ws, size_t ws_size,
                              hipStream_t stream) {
  const float* x       = (const float*)d_in[0];
  const float* W_attn  = (const float*)d_in[1];
  const float* W_proj  = (const float*)d_in[2];
  const float* cache_k = (const float*)d_in[3];
  const float* cache_v = (const float*)d_in[4];
  float* out = (float*)d_out;

  char* ws = (char*)d_ws;
  const size_t MiB = 1u << 20;
  __bf16* xbf   = (__bf16*)(ws + 0 * MiB);    // 8 MiB  (B*T, C) bf16
  __bf16* wattn = (__bf16*)(ws + 8 * MiB);    // 6 MiB  packed B-frags
  __bf16* wproj = (__bf16*)(ws + 14 * MiB);   // 2 MiB  packed B-frags
  __bf16* qbuf  = (__bf16*)(ws + 16 * MiB);   // 8 MiB  (B,H,T,D)
  __bf16* kall  = (__bf16*)(ws + 24 * MiB);   // 16 MiB (B,H,P+T,D)
  __bf16* vt    = (__bf16*)(ws + 40 * MiB);   // 16 MiB (B,H,D,P+T)
  __bf16* ybf   = (__bf16*)(ws + 56 * MiB);   // 8 MiB  (B*T, C)

  k_cvt_bf16 <<<4096, 256, 0, stream>>>(x, xbf, MROWS * CC);
  k_pack_w   <<<768,  256, 0, stream>>>(W_attn, wattn, CC, NQ);
  k_pack_w   <<<256,  256, 0, stream>>>(W_proj, wproj, CC, CC);
  k_cvt_cache<<<16384,256, 0, stream>>>(cache_k, cache_v, kall, vt);
  k_gemm_qkv <<<768,  256, 0, stream>>>(xbf, wattn, qbuf, kall, vt);
  k_attn     <<<512,  256, 0, stream>>>(qbuf, kall, vt, ybf);
  k_gemm_proj<<<256,  256, 0, stream>>>(ybf, wproj, out);
}
